// DynamicQuantizer_11897059410144
// MI455X (gfx1250) — matrix operand order repaired and checked by reading, not compiled, not run on hardware
//
#include <hip/hip_runtime.h>
#include <hip/hip_bf16.h>

typedef __attribute__((ext_vector_type(16))) _Float16     v16h;
typedef __attribute__((ext_vector_type(8)))  float        v8f;
typedef __attribute__((ext_vector_type(4)))  unsigned int v4u;
typedef __attribute__((ext_vector_type(8)))  unsigned int v8u;

#define BN_EPS 1e-5f

// ---------------------------------------------------------------------------
// Implicit-GEMM 3x3 stride-2 pad-1 conv + BN + ReLU via v_wmma_f32_16x16x32_f16
//   D[oc][pos] = sum_k W[oc][k] * patch[k][pos],  k = ic*9 + kh*3 + kw
// Workgroup = OC/16 waves; each wave computes rows [16w,16w+16) x NT 16-pos
// tiles (A-fragment reused NT times). Weight tile staged to LDS by the
// Tensor Data Mover (tensor_load_to_lds, D# per CDNA5 ISA ch.8), B tile by
// VALU im2col stores.
// ---------------------------------------------------------------------------
template<int IC, int OC, int IH, int IW, int OH, int OW, int NT,
         typename InT, typename OutT>
__global__ __launch_bounds__((OC / 16) * 32)
void conv_bn_relu_wmma(const InT* __restrict__ in, const float* __restrict__ w,
                       const float* __restrict__ gg, const float* __restrict__ bb_,
                       const float* __restrict__ mm, const float* __restrict__ vv,
                       OutT* __restrict__ out)
{
    constexpr int K     = IC * 9;
    constexpr int BLOCK = (OC / 16) * 32;
    constexpr int BPT   = (NT * 512) / BLOCK;  // B-tile elements per thread

    __shared__ float    lds_a[OC * 32];        // [oc][kk]  f32, filled by TDM
    __shared__ _Float16 lds_b[NT * 16 * 32];   // [n][kk]   f16 (transposed)

    const int t    = threadIdx.x;
    const int lane = t & 31;
    const int wave = t >> 5;
    const int moff = wave * 16;

    v8f acc[NT];
    #pragma unroll
    for (int j = 0; j < NT; ++j) acc[j] = (v8f){};

    // invariant parts of the tensor DMA descriptor (D# group1, ISA §8.4)
    const unsigned lds_base = (unsigned)(unsigned long long)(void*)lds_a;
    v8u g1;
    g1[0] = (2u << 16);                                        // data_size = 4B
    g1[1] = ((unsigned)K & 0xFFFFu) << 16;                     // tensor_dim0 lo16
    g1[2] = (((unsigned)K >> 16) & 0xFFFFu)
          | (((unsigned)OC & 0xFFFFu) << 16);                  // dim0 hi | tensor_dim1 lo
    g1[3] = (((unsigned)OC >> 16) & 0xFFFFu) | (32u << 16);    // dim1 hi | tile_dim0 = 32
    g1[4] = (unsigned)OC;                                      // tile_dim1 = OC (tile_dim2 = 0)
    g1[5] = (unsigned)K;                                       // tensor_dim0_stride lo32
    g1[6] = 0u;
    g1[7] = 0u;

    for (int k0 = 0; k0 < K; k0 += 32) {
        // ---- stage A: TDM pulls the [OC x 32] f32 weight tile into LDS ----
        if (wave == 0) {
            unsigned long long ga =
                (unsigned long long)(const void*)w + (unsigned long long)k0 * 4ull;
            v4u g0;
            g0[0] = 1u;                                        // count = 1 (valid)
            g0[1] = lds_base;                                  // lds_addr
            g0[2] = (unsigned)(ga & 0xFFFFFFFFu);              // global_addr lo
            g0[3] = (unsigned)((ga >> 32) & 0x1FFFFFFu)
                  | (2u << 30);                                // addr hi | type = 2
            asm volatile("tensor_load_to_lds %0, %1"
                         :: "s"(g0), "s"(g1) : "memory");
        }
        // ---- stage B: im2col patch with zero padding (NT*512 elems) ----
        #pragma unroll
        for (int i = 0; i < BPT; ++i) {
            int e    = t + i * BLOCK;
            int kk   = e / (NT * 16);
            int n    = e % (NT * 16);
            int k    = k0 + kk;
            int ic   = k / 9;
            int r    = k - ic * 9;
            int kh   = r / 3;
            int kw   = r - kh * 3;
            int pos  = blockIdx.x * (NT * 16) + n;
            int bidx = pos / (OH * OW);
            int rem  = pos - bidx * (OH * OW);
            int oh   = rem / OW;
            int ow   = rem - oh * OW;
            int ih   = oh * 2 + kh - 1;
            int iw   = ow * 2 + kw - 1;
            _Float16 val = (_Float16)0.f;
            if (ih >= 0 && ih < IH && iw >= 0 && iw < IW)
                val = (_Float16)in[((bidx * IC + ic) * IH + ih) * IW + iw];
            lds_b[n * 32 + kk] = val;
        }
        if (wave == 0) __builtin_amdgcn_s_wait_tensorcnt(0);
        __syncthreads();

        // ---- fragments: contiguous k order (k = 16*hb + i), same in A and B ----
        const int m  = lane & 15;
        const int hb = lane >> 4;              // half-wave select
        v16h af;
        #pragma unroll
        for (int i = 0; i < 16; ++i) {
            int kk = hb * 16 + i;                    // A: K 0-15 | 16-31
            af[i] = (_Float16)lds_a[(moff + m) * 32 + kk];
        }
        #pragma unroll
        for (int j = 0; j < NT; ++j) {
            v16h bf;
            #pragma unroll
            for (int i = 0; i < 16; ++i)
                bf[i] = lds_b[(j * 16 + m) * 32 + hb * 16 + i];
            acc[j] = __builtin_amdgcn_wmma_f32_16x16x32_f16(false, af, false, bf,
                                                            (short)0, acc[j],
                                                            false, false);
        }
        __syncthreads();
    }

    // ---- epilogue: fused BN + ReLU, scatter to NCHW ----
    const int n = lane & 15;
    #pragma unroll
    for (int r = 0; r < 8; ++r) {
        int   oc  = moff + r + (lane >> 4) * 8;  // C/D layout: VGPR r -> M=r (+8 hi half)
        float inv = gg[oc] * rsqrtf(vv[oc] + BN_EPS);
        float off = bb_[oc] - mm[oc] * inv;
        #pragma unroll
        for (int j = 0; j < NT; ++j) {
            int pos  = blockIdx.x * (NT * 16) + j * 16 + n;
            int bidx = pos / (OH * OW);
            int rem  = pos - bidx * (OH * OW);
            int oh   = rem / OW;
            int ow   = rem - oh * OW;
            float val = fmaxf(acc[j][r] * inv + off, 0.f);
            out[((bidx * OC + oc) * OH + oh) * OW + ow] = (OutT)val;
        }
    }
}

// ---------------------------------------------------------------------------
// conv3 (1x1, 32->6) + BN + spatial mean -> centers[8][6]
// ---------------------------------------------------------------------------
__global__ __launch_bounds__(256)
void conv3_bn_mean(const float* __restrict__ c2, const float* __restrict__ w3,
                   const float* __restrict__ g, const float* __restrict__ b,
                   const float* __restrict__ m, const float* __restrict__ v,
                   float* __restrict__ centers_ws, float* __restrict__ centers_out)
{
    __shared__ float red[256];
    const int bidx = blockIdx.x;
    const int t    = threadIdx.x;
    const float* src = c2 + (size_t)(bidx * 32) * 256;

    float vals[6];
    #pragma unroll
    for (int k = 0; k < 6; ++k) {
        float s = 0.f;
        #pragma unroll
        for (int ic = 0; ic < 32; ++ic)
            s += w3[k * 32 + ic] * src[ic * 256 + t];
        float inv = g[k] * rsqrtf(v[k] + BN_EPS);
        vals[k] = s * inv + (b[k] - m[k] * inv);
    }
    #pragma unroll
    for (int k = 0; k < 6; ++k) {
        red[t] = vals[k];
        __syncthreads();
        for (int s = 128; s > 0; s >>= 1) {
            if (t < s) red[t] += red[t + s];
            __syncthreads();
        }
        if (t == 0) {
            float c = red[0] * (1.f / 256.f);
            centers_ws[bidx * 6 + k]  = c;
            centers_out[bidx * 6 + k] = c;
        }
        __syncthreads();
    }
}

// ---------------------------------------------------------------------------
// Quantization: HBM-bound phase (~84 MB traffic). float4 (b128) IO.
// ---------------------------------------------------------------------------
__global__ __launch_bounds__(256)
void quantize_kernel(const float* __restrict__ x, const float* __restrict__ centers,
                     float* __restrict__ qbar, float* __restrict__ qsoft,
                     float* __restrict__ qhard, float* __restrict__ sym)
{
    __shared__ float c[6];
    const long long v4   = (long long)blockIdx.x * 256 + threadIdx.x;
    const long long base = v4 * 4;
    const int bidx = (int)(base >> 19);         // 128*64*64 = 2^19 elems per batch
    if (threadIdx.x < 6) c[threadIdx.x] = centers[bidx * 6 + threadIdx.x];
    __syncthreads();

    float4 xv = reinterpret_cast<const float4*>(x)[v4];
    float xs[4] = {xv.x, xv.y, xv.z, xv.w};
    float qb[4], qs[4], qh[4], sy[4];
    #pragma unroll
    for (int e = 0; e < 4; ++e) {
        float d[6];
        float dmin = 3.402823466e38f;
        int   amin = 0;
        #pragma unroll
        for (int k = 0; k < 6; ++k) {
            float dd = xs[e] - c[k];
            d[k] = dd * dd;
            if (d[k] < dmin) { dmin = d[k]; amin = k; }   // strict < keeps first
        }
        float se = 0.f, sc = 0.f;
        #pragma unroll
        for (int k = 0; k < 6; ++k) {
            float p = __expf(dmin - d[k]);      // SIGMA=1; max-subtracted softmax
            se += p;
            sc += p * c[k];
        }
        float soft = sc / se;
        float hard = c[amin];
        qs[e] = soft;
        qh[e] = hard;
        qb[e] = soft + (hard - soft);
        sy[e] = (float)amin;
    }
    reinterpret_cast<float4*>(qbar)[v4]  = make_float4(qb[0], qb[1], qb[2], qb[3]);
    reinterpret_cast<float4*>(qsoft)[v4] = make_float4(qs[0], qs[1], qs[2], qs[3]);
    reinterpret_cast<float4*>(qhard)[v4] = make_float4(qh[0], qh[1], qh[2], qh[3]);
    reinterpret_cast<float4*>(sym)[v4]   = make_float4(sy[0], sy[1], sy[2], sy[3]);
}

extern "C" void kernel_launch(void* const* d_in, const int* in_sizes, int n_in,
                              void* d_out, int out_size, void* d_ws, size_t ws_size,
                              hipStream_t stream)
{
    (void)in_sizes; (void)n_in; (void)out_size; (void)ws_size;

    const float* x  = (const float*)d_in[0];
    const float* cf = (const float*)d_in[1];
    const float* w1 = (const float*)d_in[2];
    const float* g1 = (const float*)d_in[3];
    const float* b1 = (const float*)d_in[4];
    const float* m1 = (const float*)d_in[5];
    const float* v1 = (const float*)d_in[6];
    const float* w2 = (const float*)d_in[7];
    const float* g2 = (const float*)d_in[8];
    const float* b2 = (const float*)d_in[9];
    const float* m2 = (const float*)d_in[10];
    const float* v2 = (const float*)d_in[11];
    const float* w3 = (const float*)d_in[12];
    const float* g3 = (const float*)d_in[13];
    const float* b3 = (const float*)d_in[14];
    const float* m3 = (const float*)d_in[15];
    const float* v3 = (const float*)d_in[16];

    // workspace layout
    char*     ws   = (char*)d_ws;
    _Float16* c1   = (_Float16*)ws;                         // 8*64*32*32 f16 = 1 MiB
    float*    c2   = (float*)(ws + 8 * 64 * 32 * 32 * 2);   // 8*32*16*16 f32 = 256 KiB
    float*    cent = (float*)(ws + 1048576 + 262144);       // 48 f32

    float* outp = (float*)d_out;
    const long long N = 8LL * 128 * 64 * 64;                // 4,194,304

    // conv1: 128->64, 64x64 -> 32x32.  8192 positions / (4 tiles*16) = 128 WGs.
    conv_bn_relu_wmma<128, 64, 64, 64, 32, 32, 4, float, _Float16>
        <<<128, 128, 0, stream>>>(cf, w1, g1, b1, m1, v1, c1);

    // conv2: 64->32, 32x32 -> 16x16.  2048 positions / 64 = 32 WGs.
    conv_bn_relu_wmma<64, 32, 32, 32, 16, 16, 4, _Float16, float>
        <<<32, 64, 0, stream>>>(c1, w2, g2, b2, m2, v2, c2);

    // conv3 + BN + mean -> centers (to ws and to tail of d_out)
    conv3_bn_mean<<<8, 256, 0, stream>>>(c2, w3, g3, b3, m3, v3, cent, outp + 4 * N);

    // quantization (memory-bound): 4096 blocks x 256 threads x 4 elems
    quantize_kernel<<<(int)(N / 1024), 256, 0, stream>>>(
        x, cent, outp, outp + N, outp + 2 * N, outp + 3 * N);
}
